// GAT_63797444215582
// MI455X (gfx1250) — matrix operand order repaired and checked by reading, not compiled, not run on hardware
//
#include <hip/hip_runtime.h>
#include <hip/hip_bf16.h>

typedef __attribute__((ext_vector_type(8)))  _Float16 v8h;
typedef __attribute__((ext_vector_type(16))) _Float16 v16h;
typedef __attribute__((ext_vector_type(8)))  float    v8f;

#define GNUM 256  // graphs

// ---------------- utility kernels ----------------

__global__ void fill_kernel(float* __restrict__ p, float v, long long n) {
    long long t = (long long)blockIdx.x * blockDim.x + threadIdx.x;
    if (t < n) p[t] = v;
}

__global__ void cvt_f16_kernel(const float* __restrict__ s, _Float16* __restrict__ d, long long n) {
    long long t = (long long)blockIdx.x * blockDim.x + threadIdx.x;
    if (t < n) d[t] = (_Float16)s[t];
}

// Pack W [di, dow] (row-major f32) into WMMA B-fragment layout:
// [kt][nt][lane 0..31][half 0..15];  lane l: N = nt*16 + (l&15), K = kt*32 + (i>>3)*16 + (l>>4)*8 + (i&7)
__global__ void pack_w_kernel(const float* __restrict__ W, _Float16* __restrict__ dst,
                              int di, int dow) {
    int t = blockIdx.x * blockDim.x + threadIdx.x;
    int total = di * dow;
    if (t >= total) return;
    int i = t & 15;
    int l = (t >> 4) & 31;
    int rest = t >> 9;                 // kt*ntiles + nt
    int ntiles = dow >> 4;
    int nt = rest % ntiles;
    int kt = rest / ntiles;
    int K = kt * 32 + ((i >> 3) << 4) + ((l >> 4) << 3) + (i & 7);
    int Ncol = nt * 16 + (l & 15);
    dst[t] = (_Float16)W[(long long)K * dow + Ncol];
}

// ---------------- WMMA GEMM: out[N,dow] = Xh[N,di] @ Wpacked + bias ----------------
// One wave per 16-row tile; loops over n-tiles (fresh v8f acc) and k-tiles.
__global__ void gemm_wmma_kernel(const _Float16* __restrict__ Xh,
                                 const _Float16* __restrict__ Wp,
                                 const float* __restrict__ bias,
                                 float* __restrict__ out,
                                 int n, int di, int dow) {
    int gtid = blockIdx.x * blockDim.x + threadIdx.x;
    int wave = gtid >> 5;
    int lane = threadIdx.x & 31;
    int mtiles = (n + 15) >> 4;
    if (wave >= mtiles) return;        // wave-uniform

    int ntiles = dow >> 4;
    int ktiles = di >> 5;
    int rowA = wave * 16 + (lane & 15);
    if (rowA > n - 1) rowA = n - 1;    // safe clamp for A loads
    int kbase = (lane >> 4) * 8;       // A layout: lanes<16 -> K 0..7 & 16..23, else 8..15 & 24..31

    for (int nt = 0; nt < ntiles; ++nt) {
        v8f acc = {};
        for (int kt = 0; kt < ktiles; ++kt) {
            int k0 = kt * 32 + kbase;
            const _Float16* ap = Xh + (long long)rowA * di + k0;
            v8h alo = *(const v8h*)(ap);
            v8h ahi = *(const v8h*)(ap + 16);
            v16h a = __builtin_shufflevector(alo, ahi,
                       0,1,2,3,4,5,6,7,8,9,10,11,12,13,14,15);
            v16h b = *(const v16h*)(Wp + (((long long)(kt * ntiles + nt)) * 32 + lane) * 16);
            acc = __builtin_amdgcn_wmma_f32_16x16x32_f16(
                      false, a, false, b, (short)0, acc, false, false);
        }
        int ncol = nt * 16 + (lane & 15);
        float bv = bias[ncol];
        int mrow0 = wave * 16 + ((lane >> 4) << 3);   // +0 for lanes<16, +8 else
        #pragma unroll
        for (int r = 0; r < 8; ++r) {
            int row = mrow0 + r;
            if (row < n) out[(long long)row * dow + ncol] = acc[r] + bv;
        }
    }
}

// ---------------- edge kernels ----------------

__device__ inline void atomicMaxFloat(float* addr, float value) {
    if (value >= 0.0f)
        atomicMax((int*)addr, __float_as_int(value));
    else
        atomicMin((unsigned int*)addr, __float_as_uint(value));
}

// e = a . leakyrelu(xl[src] + xr[dst]);  store e; atomic max into emax[dst]
__global__ void edge_score_kernel(const int* __restrict__ ei, int E, int Nn,
                                  const float* __restrict__ XL, const float* __restrict__ XR,
                                  const float* __restrict__ avec, int dow,
                                  float* __restrict__ ebuf, float* __restrict__ emax) {
    int eid = blockIdx.x * blockDim.x + threadIdx.x;
    int Etot = E + Nn;
    if (eid >= Etot) return;
    int s, d;
    if (eid < E) { s = ei[eid]; d = ei[E + eid]; }
    else         { s = eid - E; d = s; }
    const float* xl = XL + (long long)s * dow;
    const float* xr = XR + (long long)d * dow;
    float e = 0.0f;
    for (int k = 0; k < dow; ++k) {
        float z = xl[k] + xr[k];
        z = z > 0.0f ? z : 0.2f * z;
        e += z * avec[k];
    }
    ebuf[eid] = e;
    atomicMaxFloat(&emax[d], e);
}

// ee = exp(e - emax[dst]); store ee; atomic add into denom[dst]
__global__ void edge_exp_kernel(const int* __restrict__ ei, int E, int Nn,
                                const float* __restrict__ emax,
                                float* __restrict__ ebuf, float* __restrict__ denom) {
    int eid = blockIdx.x * blockDim.x + threadIdx.x;
    int Etot = E + Nn;
    if (eid >= Etot) return;
    int d = (eid < E) ? ei[E + eid] : (eid - E);
    float ee = __expf(ebuf[eid] - emax[d]);
    ebuf[eid] = ee;
    atomicAdd(&denom[d], ee);
}

// out[dst] += (ee/denom[dst]) * xl[src];  one thread per (edge, 4 columns)
__global__ void edge_agg_kernel(const int* __restrict__ ei, int E, int Nn,
                                const float* __restrict__ XL,
                                const float* __restrict__ ebuf, const float* __restrict__ denom,
                                float* __restrict__ H, int dow) {
    int lpe = dow >> 2;
    long long t = (long long)blockIdx.x * blockDim.x + threadIdx.x;
    long long total = (long long)(E + Nn) * lpe;
    if (t >= total) return;
    int eid = (int)(t / lpe);
    int sub = (int)(t % lpe);
    int s, d;
    if (eid < E) { s = ei[eid]; d = ei[E + eid]; }
    else         { s = eid - E; d = s; }
    float alpha = ebuf[eid] / denom[d];
    int c0 = sub * 4;
    const float* xl = XL + (long long)s * dow + c0;
    float* hp = H + (long long)d * dow + c0;
    #pragma unroll
    for (int j = 0; j < 4; ++j)
        atomicAdd(&hp[j], alpha * xl[j]);
}

// ---------------- fused bias+ReLU+BatchNorm (training-mode stats), in-place OK ----------------
__global__ void bn_kernel(const float* __restrict__ X, float* __restrict__ Y,
                          const float* __restrict__ bc,
                          const float* __restrict__ g, const float* __restrict__ be,
                          int n, int d, int do_relu) {
    int c = blockIdx.x;
    __shared__ float ssum[256];
    __shared__ float ssq[256];
    __shared__ float s_m, s_inv;
    float bcv = bc ? bc[c] : 0.0f;
    float s = 0.0f, q = 0.0f;
    for (int r = threadIdx.x; r < n; r += blockDim.x) {
        float v = X[(long long)r * d + c] + bcv;
        if (do_relu) v = v > 0.0f ? v : 0.0f;
        s += v; q += v * v;
    }
    ssum[threadIdx.x] = s; ssq[threadIdx.x] = q;
    __syncthreads();
    for (int off = 128; off > 0; off >>= 1) {
        if (threadIdx.x < off) {
            ssum[threadIdx.x] += ssum[threadIdx.x + off];
            ssq[threadIdx.x]  += ssq[threadIdx.x + off];
        }
        __syncthreads();
    }
    if (threadIdx.x == 0) {
        float m = ssum[0] / n;
        float v = ssq[0] / n - m * m;
        s_m = m;
        s_inv = rsqrtf(v + 1e-5f);
    }
    __syncthreads();
    float m = s_m, inv = s_inv, gc = g[c], bec = be[c];
    for (int r = threadIdx.x; r < n; r += blockDim.x) {
        float v = X[(long long)r * d + c] + bcv;
        if (do_relu) v = v > 0.0f ? v : 0.0f;
        Y[(long long)r * d + c] = gc * (v - m) * inv + bec;
    }
}

// ---------------- pooling + head ----------------

__global__ void pool_kernel(const float* __restrict__ H, const int* __restrict__ batch,
                            float* __restrict__ pool, int n, int d, int off) {
    long long t = (long long)blockIdx.x * blockDim.x + threadIdx.x;
    if (t >= (long long)n * d) return;
    int node = (int)(t / d), col = (int)(t % d);
    atomicAdd(&pool[(long long)batch[node] * 256 + off + col], H[t]);
}

__global__ void head1_kernel(const float* __restrict__ pool,
                             const float* __restrict__ W5, const float* __restrict__ b5,
                             float* __restrict__ t1) {
    int t = blockIdx.x * blockDim.x + threadIdx.x;
    if (t >= GNUM * 128) return;
    int g = t / 128, j = t % 128;
    float acc = b5[j];
    const float* pr = pool + (long long)g * 256;
    for (int k = 0; k < 256; ++k) acc += pr[k] * W5[k * 128 + j];
    t1[t] = acc > 0.0f ? acc : 0.0f;
}

__global__ void head2_kernel(const float* __restrict__ t1,
                             const float* __restrict__ W6, const float* __restrict__ b6,
                             float* __restrict__ lg) {
    int t = blockIdx.x * blockDim.x + threadIdx.x;
    if (t >= GNUM * 10) return;
    int g = t / 10, j = t % 10;
    float acc = b6[j];
    const float* tr = t1 + (long long)g * 128;
    for (int k = 0; k < 128; ++k) acc += tr[k] * W6[k * 10 + j];
    lg[t] = acc;
}

__global__ void head_out_kernel(const float* __restrict__ lg, float* __restrict__ out) {
    int g = blockIdx.x * blockDim.x + threadIdx.x;
    if (g >= GNUM) return;
    float l[10];
    float m = -3.0e38f;
    for (int j = 0; j < 10; ++j) { l[j] = lg[g * 10 + j]; m = fmaxf(m, l[j]); }
    float s = 0.0f;
    for (int j = 0; j < 10; ++j) s += __expf(l[j] - m);
    float lse = __logf(s) + m;
    for (int j = 0; j < 10; ++j) {
        out[g * 10 + j]              = 1.0f / (1.0f + __expf(-l[j]));   // sigmoid
        out[GNUM * 10 + g * 10 + j]  = l[j] - lse;                      // log_softmax
    }
}

// ---------------- host orchestration ----------------

static inline int cdiv(long long a, long long b) { return (int)((a + b - 1) / b); }

extern "C" void kernel_launch(void* const* d_in, const int* in_sizes, int n_in,
                              void* d_out, int out_size, void* d_ws, size_t ws_size,
                              hipStream_t stream) {
    const float* x   = (const float*)d_in[0];
    const int*   ei  = (const int*)d_in[1];
    const int*   bat = (const int*)d_in[2];
    const int Nn = in_sizes[0] / 128;
    const int E  = in_sizes[1] / 2;
    const int Etot = E + Nn;

    // per-layer params: layer i base index = 4 + (i-1)*8 : Wl, bl, Wr, br, a, bc, g, be
    const float* W5  = (const float*)d_in[36];
    const float* b5  = (const float*)d_in[37];
    const float* g5  = (const float*)d_in[38];
    const float* be5 = (const float*)d_in[39];
    const float* W6  = (const float*)d_in[40];
    const float* b6  = (const float*)d_in[41];

    float* ws = (float*)d_ws;
    size_t o = 0;
    float* H1 = ws + o;  o += (size_t)Nn * 128;
    float* H2 = ws + o;  o += (size_t)Nn * 64;
    float* H3 = ws + o;  o += (size_t)Nn * 32;
    float* XL = ws + o;  o += (size_t)Nn * 128;
    float* XR = ws + o;  o += (size_t)Nn * 128;
    _Float16* XH  = (_Float16*)(ws + o); o += (size_t)Nn * 64;   // N*128 halfs
    _Float16* WLP = (_Float16*)(ws + o); o += 8192;              // 128*128 halfs
    _Float16* WRP = (_Float16*)(ws + o); o += 8192;
    float* EBUF = ws + o; o += (size_t)Etot;
    float* EMAX = ws + o; o += (size_t)Nn;
    float* DEN  = ws + o; o += (size_t)Nn;
    float* POOL = ws + o; o += (size_t)GNUM * 256;
    float* T1   = ws + o; o += (size_t)GNUM * 128;
    float* LG   = ws + o; o += (size_t)GNUM * 10;

    const int BS = 256;
    const int dims_in[3]  = {128, 128, 64};
    const int dims_out[3] = {128, 64, 32};
    const float* feat_in[3] = {x, H1, H2};
    float* feat_out[3] = {H1, H2, H3};

    for (int L = 0; L < 3; ++L) {
        int di = dims_in[L], dow = dims_out[L];
        int pbase = 4 + L * 8;
        const float* Wl = (const float*)d_in[pbase + 0];
        const float* bl = (const float*)d_in[pbase + 1];
        const float* Wr = (const float*)d_in[pbase + 2];
        const float* br = (const float*)d_in[pbase + 3];
        const float* av = (const float*)d_in[pbase + 4];
        const float* bc = (const float*)d_in[pbase + 5];
        const float* gg = (const float*)d_in[pbase + 6];
        const float* be = (const float*)d_in[pbase + 7];
        float* Hout = feat_out[L];

        // features -> f16, weights -> packed B-fragments
        cvt_f16_kernel<<<cdiv((long long)Nn * di, BS), BS, 0, stream>>>(feat_in[L], XH, (long long)Nn * di);
        pack_w_kernel<<<cdiv(di * dow, BS), BS, 0, stream>>>(Wl, WLP, di, dow);
        pack_w_kernel<<<cdiv(di * dow, BS), BS, 0, stream>>>(Wr, WRP, di, dow);

        // WMMA GEMMs: XL = Xh@Wl + bl, XR = Xh@Wr + br
        int mtiles = (Nn + 15) / 16;
        int gblocks = cdiv((long long)mtiles * 32, BS);
        gemm_wmma_kernel<<<gblocks, BS, 0, stream>>>(XH, WLP, bl, XL, Nn, di, dow);
        gemm_wmma_kernel<<<gblocks, BS, 0, stream>>>(XH, WRP, br, XR, Nn, di, dow);

        // softmax-aggregate over edges
        fill_kernel<<<cdiv(Nn, BS), BS, 0, stream>>>(EMAX, -__builtin_inff(), Nn);
        fill_kernel<<<cdiv(Nn, BS), BS, 0, stream>>>(DEN, 0.0f, Nn);
        fill_kernel<<<cdiv((long long)Nn * dow, BS), BS, 0, stream>>>(Hout, 0.0f, (long long)Nn * dow);
        edge_score_kernel<<<cdiv(Etot, BS), BS, 0, stream>>>(ei, E, Nn, XL, XR, av, dow, EBUF, EMAX);
        edge_exp_kernel<<<cdiv(Etot, BS), BS, 0, stream>>>(ei, E, Nn, EMAX, EBUF, DEN);
        edge_agg_kernel<<<cdiv((long long)Etot * (dow / 4), BS), BS, 0, stream>>>(
            ei, E, Nn, XL, EBUF, DEN, Hout, dow);

        // +bc, ReLU, BatchNorm (in place)
        bn_kernel<<<dow, BS, 0, stream>>>(Hout, Hout, bc, gg, be, Nn, dow, 1);
    }

    // pooling: [p1 | p2 | p3 | p4==p3]
    fill_kernel<<<cdiv((long long)GNUM * 256, BS), BS, 0, stream>>>(POOL, 0.0f, (long long)GNUM * 256);
    pool_kernel<<<cdiv((long long)Nn * 128, BS), BS, 0, stream>>>(H1, bat, POOL, Nn, 128, 0);
    pool_kernel<<<cdiv((long long)Nn * 64,  BS), BS, 0, stream>>>(H2, bat, POOL, Nn, 64, 128);
    pool_kernel<<<cdiv((long long)Nn * 32,  BS), BS, 0, stream>>>(H3, bat, POOL, Nn, 32, 192);
    pool_kernel<<<cdiv((long long)Nn * 32,  BS), BS, 0, stream>>>(H3, bat, POOL, Nn, 32, 224);

    // head
    head1_kernel<<<cdiv(GNUM * 128, BS), BS, 0, stream>>>(POOL, W5, b5, T1);
    bn_kernel<<<128, BS, 0, stream>>>(T1, T1, nullptr, g5, be5, GNUM, 128, 0);
    head2_kernel<<<cdiv(GNUM * 10, BS), BS, 0, stream>>>(T1, W6, b6, LG);
    head_out_kernel<<<cdiv(GNUM, BS), BS, 0, stream>>>(LG, (float*)d_out);
}
